// HTP_75917841924663
// MI455X (gfx1250) — matrix operand order repaired and checked by reading, not compiled, not run on hardware
//
#include <hip/hip_runtime.h>
#include <hip/hip_bf16.h>
#include <math.h>

// ---------------------------------------------------------------------------
// CDNA5 (gfx1250): wave32 + V_WMMA_F32_16X16X32_F16 + TDM (tensor_load_to_lds)
// ---------------------------------------------------------------------------

typedef __attribute__((ext_vector_type(16))) _Float16 v16h;
typedef __attribute__((ext_vector_type(8)))  float    v8f;
typedef __attribute__((ext_vector_type(4)))  unsigned int v4u;
typedef __attribute__((ext_vector_type(8)))  int      v8i_t;
typedef __attribute__((ext_vector_type(4)))  int      v4i_t;

#define WMMA_F16(a, b, c) \
  __builtin_amdgcn_wmma_f32_16x16x32_f16(false, (a), false, (b), (short)0, (c), false, false)

// Problem constants (from the reference).
constexpr int U_  = 50000, I_ = 30000, C_ = 1000;
constexpr int YN_ = 20, MN_ = 13, DN_ = 32;
constexpr int H_  = 128, L_ = 200, B_ = 256, NL_ = 3;
constexpr int NUI = U_ + I_;     // 80000 NGCF nodes (user-item graph)
constexpr int NUC = U_ + C_;     // 51000 NGCF nodes (user-cate graph)
constexpr int NNZ_UI_ = 800000, NNZ_UC_ = 200000;
constexpr float SQRTH = 11.313708498984760f;   // sqrt(128)
constexpr int WHH_HALVES = 3 * H_ * H_;        // 49152 halves = 96KB

// ---------------------------------------------------------------------------
// WMMA fragment loaders (exact gfx1250 16-bit layouts)
// ---------------------------------------------------------------------------

// A (16x32, f16) from f32 source: lane m=lane&15 holds row m. Half-wave split:
//   lanes 0-15 : elems 0-7 -> K=k0+0..7,  elems 8-15 -> K=k0+16..23
//   lanes16-31 : elems 0-7 -> K=k0+8..15, elems 8-15 -> K=k0+24..31
__device__ inline v16h loadA_f32(const float* __restrict__ src, int ld, int k0,
                                 int valid_rows) {
  const int lane = threadIdx.x & 31;
  const int m    = lane & 15;
  const int hi   = (lane >> 4) & 1;
  v16h a;
  if (m < valid_rows) {
    const float* p = src + (size_t)m * ld + k0 + hi * 8;
#pragma unroll
    for (int e = 0; e < 8; ++e) {
      a[e]     = (_Float16)p[e];
      a[e + 8] = (_Float16)p[e + 16];
    }
  } else {
#pragma unroll
    for (int e = 0; e < 16; ++e) a[e] = (_Float16)0.0f;
  }
  return a;
}

// A fragment of an elementwise product x*y (NGCF bilinear term).
__device__ inline v16h loadA_prod_f32(const float* __restrict__ x,
                                      const float* __restrict__ y, int ld,
                                      int k0, int valid_rows) {
  const int lane = threadIdx.x & 31;
  const int m    = lane & 15;
  const int hi   = (lane >> 4) & 1;
  v16h a;
  if (m < valid_rows) {
    const float* px = x + (size_t)m * ld + k0 + hi * 8;
    const float* py = y + (size_t)m * ld + k0 + hi * 8;
#pragma unroll
    for (int e = 0; e < 8; ++e) {
      a[e]     = (_Float16)(px[e] * py[e]);
      a[e + 8] = (_Float16)(px[e + 16] * py[e + 16]);
    }
  } else {
#pragma unroll
    for (int e = 0; e < 16; ++e) a[e] = (_Float16)0.0f;
  }
  return a;
}

// B (32x16, f16) for out = X @ W^T with W pre-converted to f16, row-major
// [O][K]: B[k,o] = W[o,k]. Lane holds column o = o0+(lane&15); elems 0-7 ->
// K=k0+hi*8+0..7, elems 8-15 -> K=k0+16+hi*8+0..7 (same split as A).
// Works for global or LDS (generic) pointers.
__device__ inline v16h loadB_Wt_f16(const _Float16* __restrict__ W, int ld,
                                    int o0, int k0) {
  const int lane = threadIdx.x & 31;
  const int o    = o0 + (lane & 15);
  const _Float16* p = W + (size_t)o * ld + k0 + ((lane >> 4) & 1) * 8;
  v16h b;
#pragma unroll
  for (int e = 0; e < 8; ++e) {
    b[e]     = p[e];
    b[e + 8] = p[e + 16];
  }
  return b;
}

__device__ inline v8f zero_v8f() {
  v8f d;
#pragma unroll
  for (int i = 0; i < 8; ++i) d[i] = 0.0f;
  return d;
}

__device__ inline float wave_sum(float v) {
#pragma unroll
  for (int off = 16; off > 0; off >>= 1) v += __shfl_xor(v, off, 32);
  return v;
}

__device__ inline float sigmoidf(float x) { return 1.0f / (1.0f + expf(-x)); }

// ---------------------------------------------------------------------------
// TDM: load Whh (f16, 49152 halves = 96KB) from global into LDS as a 1-D
// tensor tile via the Tensor Data Mover. D# per CDNA5 ISA:
//   group0: count=1 | lds_addr | global_addr | type=2
//   group1: data_size=1(2B), tensor_dim0=tile_dim0=49152, dims1=1, stride=dim0
// ---------------------------------------------------------------------------
__device__ inline void load_whh_to_lds(const _Float16* __restrict__ g,
                                       _Float16* l) {
#if defined(__gfx1250__) && __has_builtin(__builtin_amdgcn_tensor_load_to_lds)
  unsigned lds_off = (unsigned)(uintptr_t)l;  // addr[31:0] == LDS byte offset
  unsigned long long ga = (unsigned long long)(uintptr_t)g;
  v4u g0;
  g0[0] = 1u;                                   // count=1, user mode
  g0[1] = lds_off;                              // lds_addr [63:32]
  g0[2] = (unsigned)(ga & 0xffffffffu);         // global_addr lo
  g0[3] = (unsigned)((ga >> 32) & 0x01ffffffu)  // global_addr hi [120:96]
          | (2u << 30);                         // type=2 ("image")
  v8i_t g1;
  g1[0] = (int)(1u << 16);            // workgroup_mask=0, data_size=1 (2B)
  g1[1] = (int)(0xC000u << 16);       // tensor_dim0[15:0]=49152 @ bits[63:48]
  g1[2] = (int)(1u << 16);            // tensor_dim0 hi=0 | tensor_dim1 lo=1
  g1[3] = (int)(0xC000u << 16);       // tensor_dim1 hi=0 | tile_dim0=49152
  g1[4] = 1;                          // tile_dim1=1, tile_dim2=0
  g1[5] = (int)49152;                 // tensor_dim0_stride lo32
  g1[6] = 0;                          // stride hi | dim1_stride lo
  g1[7] = 0;
  v4i_t z;
  z[0] = 0; z[1] = 0; z[2] = 0; z[3] = 0;
#if defined(__clang_major__) && (__clang_major__ >= 23)
  v8i_t z8;
#pragma unroll
  for (int i = 0; i < 8; ++i) z8[i] = 0;
  __builtin_amdgcn_tensor_load_to_lds(g0, g1, z, z, z8, 0);
#else
  __builtin_amdgcn_tensor_load_to_lds(g0, g1, z, z, 0);
#endif
  __builtin_amdgcn_s_wait_tensorcnt(0);
#else
  // Fallback: manual wave copy (16B chunks).
  const uint4* src = (const uint4*)g;
  uint4* dst = (uint4*)l;
  for (int i = threadIdx.x & 31; i < WHH_HALVES * 2 / 16; i += 32)
    dst[i] = src[i];
#endif
}

// ---------------------------------------------------------------------------
// Utility kernels
// ---------------------------------------------------------------------------

__global__ __launch_bounds__(256) void zero_kernel(float* __restrict__ p,
                                                   long long n) {
  long long i = (long long)blockIdx.x * 256 + threadIdx.x;
  if (i < n) p[i] = 0.0f;
}

__global__ __launch_bounds__(256) void cvt_f16_kernel(
    const float* __restrict__ s, _Float16* __restrict__ d, long long n) {
  long long i = (long long)blockIdx.x * 256 + threadIdx.x;
  if (i < n) d[i] = (_Float16)s[i];
}

__global__ __launch_bounds__(256) void scale_kernel(const float* __restrict__ s,
                                                    float* __restrict__ d,
                                                    long long n, float k) {
  long long i = (long long)blockIdx.x * 256 + threadIdx.x;
  if (i < n) d[i] = s[i] * k;
}

// ego = concat(a[na,H], b[nb,H]); acc = ego  (embs[0] term of the NGCF mean)
__global__ __launch_bounds__(256) void concat2_kernel(
    const float* __restrict__ a, int na, const float* __restrict__ b, int nb,
    float* __restrict__ ego, float* __restrict__ acc) {
  long long i = (long long)blockIdx.x * 256 + threadIdx.x;
  long long tot = (long long)(na + nb) * H_;
  if (i < tot) {
    long long row = i >> 7;
    int h = (int)(i & 127);
    float v = (row < (long long)na) ? a[row * H_ + h]
                                    : b[(row - na) * H_ + h];
    ego[i] = v;
    acc[i] = v;
  }
}

// ---------------------------------------------------------------------------
// NGCF sparse scatter: side[rows[e],:] += vals[e] * ego[cols[e],:]
// One wave per edge (lane j owns 4 floats): float4 gather + 4 f32 atomics.
// ego/side fit in the 192MB L2 -> gathers and atomics are L2-resident.
// ---------------------------------------------------------------------------
__global__ __launch_bounds__(256) void spmm_scatter_kernel(
    const int* __restrict__ rows, const int* __restrict__ cols,
    const float* __restrict__ vals, const float* __restrict__ ego,
    float* __restrict__ side, int nnz) {
  long long idx = (long long)blockIdx.x * 256 + threadIdx.x;
  int e = (int)(idx >> 5);
  int j = (int)(idx & 31);
  if (e < nnz) {
    int c = cols[e];
    int r = rows[e];
    float v = vals[e];
    const float4 eg = *(const float4*)(ego + (size_t)c * H_ + j * 4);
    float* sp = side + (size_t)r * H_ + j * 4;
    atomicAdd(sp + 0, v * eg.x);
    atomicAdd(sp + 1, v * eg.y);
    atomicAdd(sp + 2, v * eg.z);
    atomicAdd(sp + 3, v * eg.w);
  }
}

// ---------------------------------------------------------------------------
// NGCF dense layer (per 16-row tile, 1 wave):
//   sum = lrelu(side @ Wgc^T + bgc); bi = lrelu((ego*side) @ Wbi^T + bbi)
//   ego' = sum + bi;  acc += ego'/max(||ego'||,1e-12)
// 64 v_wmma_f32_16x16x32_f16 per wave. Weights pre-converted to f16.
// In-place ego update safe: each block touches only its own 16 rows.
// ---------------------------------------------------------------------------
__global__ __launch_bounds__(32) void ngcf_dense_kernel(
    const float* __restrict__ ego, const float* __restrict__ side,
    float* __restrict__ ego_out, float* __restrict__ acc,
    const _Float16* __restrict__ Wgc, const float* __restrict__ bgc,
    const _Float16* __restrict__ Wbi, const float* __restrict__ bbi, int N) {
  __shared__ float tile[16][H_ + 1];
  __shared__ float rnorm[16];

  const int lane  = threadIdx.x & 31;
  const int row0  = blockIdx.x * 16;
  const int rem   = N - row0;
  const int valid = rem < 16 ? rem : 16;
  const float* egoT  = ego + (size_t)row0 * H_;
  const float* sideT = side + (size_t)row0 * H_;

  v16h aS[4], aP[4];
#pragma unroll
  for (int kc = 0; kc < 4; ++kc) {
    aS[kc] = loadA_f32(sideT, H_, kc * 32, valid);
    aP[kc] = loadA_prod_f32(egoT, sideT, H_, kc * 32, valid);
  }

  for (int ot = 0; ot < 8; ++ot) {
    v8f dS = zero_v8f();
    v8f dP = zero_v8f();
#pragma unroll
    for (int kc = 0; kc < 4; ++kc) {
      dS = WMMA_F16(aS[kc], loadB_Wt_f16(Wgc, H_, ot * 16, kc * 32), dS);
      dP = WMMA_F16(aP[kc], loadB_Wt_f16(Wbi, H_, ot * 16, kc * 32), dP);
    }
    const int n  = ot * 16 + (lane & 15);
    const int mb = (lane >> 4) * 8;
    const float bg = bgc[n];
    const float bb = bbi[n];
#pragma unroll
    for (int v = 0; v < 8; ++v) {
      float s = dS[v] + bg;  s = s > 0.0f ? s : 0.01f * s;   // leaky_relu
      float p = dP[v] + bb;  p = p > 0.0f ? p : 0.01f * p;
      tile[mb + v][n] = s + p;
    }
  }
  __syncthreads();

  if (lane < 16) {
    float ss = 0.0f;
#pragma unroll 4
    for (int h = 0; h < H_; ++h) {
      float x = tile[lane][h];
      ss += x * x;
    }
    rnorm[lane] = 1.0f / fmaxf(sqrtf(ss), 1e-12f);
  }
  __syncthreads();

  for (int idx = lane; idx < 16 * H_; idx += 32) {
    int m = idx >> 7, h = idx & 127;
    if (m < valid) {
      float x = tile[m][h];
      size_t o = (size_t)(row0 + m) * H_ + h;
      ego_out[o] = x;
      acc[o] += x * rnorm[m];
    }
  }
}

// item_time[i,:] = mean of year/month/day embeddings (it_vals == 1/3)
__global__ __launch_bounds__(256) void item_time_kernel(
    const int* __restrict__ it_cols, const float* __restrict__ it_vals,
    const float* __restrict__ ye, const float* __restrict__ me,
    const float* __restrict__ de, float* __restrict__ item_time) {
  long long i = (long long)blockIdx.x * 256 + threadIdx.x;
  if (i < (long long)I_ * H_) {
    int item = (int)(i >> 7), h = (int)(i & 127);
    float s = 0.0f;
#pragma unroll
    for (int j = 0; j < 3; ++j) {
      int c = it_cols[item * 3 + j];
      float v = it_vals[item * 3 + j];
      const float* tab;
      int cc;
      if (c < YN_)            { tab = ye; cc = c; }
      else if (c < YN_ + MN_) { tab = me; cc = c - YN_; }
      else                    { tab = de; cc = c - YN_ - MN_; }
      s += v * tab[(size_t)cc * H_ + h];
    }
    item_time[i] = s;
  }
}

// seqs[b,l,:] = mask ? 0 : items_emb[id]*sqrt(H) + abs_pos[l] + item_time[id]
__global__ __launch_bounds__(256) void seqs_kernel(
    const int* __restrict__ log_seqs, const float* __restrict__ items_emb,
    const float* __restrict__ abs_pos, const float* __restrict__ item_time,
    float* __restrict__ seqs) {
  long long i = (long long)blockIdx.x * 256 + threadIdx.x;
  if (i < (long long)B_ * L_ * H_) {
    int h = (int)(i & 127);
    long long bl = i >> 7;
    int l = (int)(bl % L_);
    int id = log_seqs[bl];
    float v = 0.0f;
    if (id != 0) {
      v = items_emb[(size_t)id * H_ + h] * SQRTH + abs_pos[(size_t)l * H_ + h] +
          item_time[(size_t)id * H_ + h];
    }
    seqs[i] = v;
  }
}

// fi == 1/L exactly (softmax over a broadcast-constant axis), so
// E row = LN(mean_l seqs[b,l,:]); the big attention GEMMs are dead code.
__global__ __launch_bounds__(256) void seq_mean_kernel(
    const float* __restrict__ seqs, float* __restrict__ seq_mean) {
  int i = blockIdx.x * 256 + threadIdx.x;
  if (i < B_ * H_) {
    int b = i >> 7, h = i & 127;
    float s = 0.0f;
    for (int l = 0; l < L_; ++l) s += seqs[((size_t)b * L_ + l) * H_ + h];
    seq_mean[i] = s * (1.0f / L_);
  }
}

// Row-wise LayerNorm (one wave per row, shuffle reductions).
__global__ __launch_bounds__(32) void ln_rows_kernel(
    const float* __restrict__ x, const float* __restrict__ g,
    const float* __restrict__ bta, float* __restrict__ y, int nrows) {
  int row = blockIdx.x;
  if (row >= nrows) return;
  const int lane = threadIdx.x;
  const float* xr = x + (size_t)row * H_;
  float v[4], s = 0.0f;
#pragma unroll
  for (int j = 0; j < 4; ++j) { v[j] = xr[lane + 32 * j]; s += v[j]; }
  float mean = wave_sum(s) * (1.0f / H_);
  float vs = 0.0f;
#pragma unroll
  for (int j = 0; j < 4; ++j) { float d = v[j] - mean; vs += d * d; }
  float inv = rsqrtf(wave_sum(vs) * (1.0f / H_) + 1e-8f);
#pragma unroll
  for (int j = 0; j < 4; ++j) {
    int h = lane + 32 * j;
    y[(size_t)row * H_ + h] = (v[j] - mean) * inv * g[h] + bta[h];
  }
}

// xp = seqs @ Wih^T + bih : [51200,128] @ [128,384], f16 weights.
__global__ __launch_bounds__(32) void xp_gemm_kernel(
    const float* __restrict__ seqs, const _Float16* __restrict__ Wih,
    const float* __restrict__ bih, float* __restrict__ xp) {
  const int lane = threadIdx.x & 31;
  const int row0 = blockIdx.x * 16;
  const float* At = seqs + (size_t)row0 * H_;
  v16h a[4];
#pragma unroll
  for (int kc = 0; kc < 4; ++kc) a[kc] = loadA_f32(At, H_, kc * 32, 16);
  for (int ot = 0; ot < 24; ++ot) {
    v8f d = zero_v8f();
#pragma unroll
    for (int kc = 0; kc < 4; ++kc)
      d = WMMA_F16(a[kc], loadB_Wt_f16(Wih, H_, ot * 16, kc * 32), d);
    const int n = ot * 16 + (lane & 15);
    const int mb = (lane >> 4) * 8;
    const float bias = bih[n];
#pragma unroll
    for (int v = 0; v < 8; ++v)
      xp[(size_t)(row0 + mb + v) * 384 + n] = d[v] + bias;
  }
}

// ---------------------------------------------------------------------------
// GRU scan: 16 persistent blocks (B=256 / 16 rows each) -> the recurrence
// needs NO cross-block sync. Whh (f16, 96KB) is staged into LDS once per
// block via the Tensor Data Mover; B-fragments then come from ds_load.
// Dynamic LDS: 96KB Whh + 8.25KB h + 24KB hp = ~129KB (of 320KB/WGP).
// ---------------------------------------------------------------------------
__global__ __launch_bounds__(32) void gru_scan_kernel(
    const float* __restrict__ xp, const _Float16* __restrict__ Whh16,
    const float* __restrict__ bhh, float* __restrict__ Fu) {
  extern __shared__ char smem[];
  _Float16* WhhS = (_Float16*)smem;                               // 98304 B
  float* h_s  = (float*)(smem + WHH_HALVES * 2);                  // [16][132]
  float* hp_s = (float*)(smem + WHH_HALVES * 2 + 16 * 132 * 4);   // [16][384]
  const int lane = threadIdx.x & 31;
  const int r0 = blockIdx.x * 16;
  const int LDH = H_ + 4;   // padded ld for conflict-free A loads

  load_whh_to_lds(Whh16, WhhS);
  for (int idx = lane; idx < 16 * LDH; idx += 32) h_s[idx] = 0.0f;
  __syncthreads();

  for (int t = 0; t < L_; ++t) {
    v16h a[4];
#pragma unroll
    for (int kc = 0; kc < 4; ++kc) a[kc] = loadA_f32(h_s, LDH, kc * 32, 16);
    for (int ot = 0; ot < 24; ++ot) {
      v8f d = zero_v8f();
#pragma unroll
      for (int kc = 0; kc < 4; ++kc)
        d = WMMA_F16(a[kc], loadB_Wt_f16(WhhS, H_, ot * 16, kc * 32), d);
      const int n = ot * 16 + (lane & 15);
      const int mb = (lane >> 4) * 8;
      const float bias = bhh[n];
#pragma unroll
      for (int v = 0; v < 8; ++v) hp_s[(mb + v) * 384 + n] = d[v] + bias;
    }
    __syncthreads();
    for (int idx = lane; idx < 16 * H_; idx += 32) {
      int m = idx >> 7, hh = idx & 127;
      const float* xpt = xp + ((size_t)(r0 + m) * L_ + t) * 384;
      float r = sigmoidf(xpt[hh] + hp_s[m * 384 + hh]);
      float z = sigmoidf(xpt[H_ + hh] + hp_s[m * 384 + H_ + hh]);
      float cand = tanhf(xpt[2 * H_ + hh] + r * hp_s[m * 384 + 2 * H_ + hh]);
      float hprev = h_s[m * LDH + hh];
      float hnew = (1.0f - z) * cand + z * hprev;
      h_s[m * LDH + hh] = hnew;
      Fu[((size_t)(r0 + m) * L_ + t) * H_ + hh] = hnew;
    }
    __syncthreads();
  }
}

__global__ __launch_bounds__(128) void flmean_kernel(
    const float* __restrict__ Fu, float* __restrict__ flm) {
  int h = threadIdx.x;
  if (h < H_) {
    float s = 0.0f;
    for (int b = 0; b < B_; ++b) s += Fu[((size_t)b * L_ + (L_ - 1)) * H_ + h];
    flm[h] = s * (1.0f / B_);
  }
}

// con1: mean over (B,H). JAX PRNG perms irreproducible on-device ->
// deterministic modular bijections.
__global__ __launch_bounds__(256) void con1_kernel(
    const float* __restrict__ user_emb_ngcf, const int* __restrict__ user_ids,
    const float* __restrict__ flm, float* __restrict__ acc) {
  __shared__ float red[256];
  int i = blockIdx.x * 256 + threadIdx.x;
  float val = 0.0f;
  if (i < B_ * H_) {
    int b = i >> 7, h = i & 127;
    int pb = (b * 77 + 13) & 255;
    int ph = (h * 45 + 9) & 127;
    float pos = user_emb_ngcf[(size_t)user_ids[b] * H_ + h] * flm[h];
    float neg = user_emb_ngcf[(size_t)user_ids[pb] * H_ + ph] * flm[h];
    val = -logf(1e-8f + sigmoidf(pos)) - logf(1e-8f + 1.0f - sigmoidf(neg));
  }
  red[threadIdx.x] = val;
  __syncthreads();
  for (int s = 128; s > 0; s >>= 1) {
    if (threadIdx.x < s) red[threadIdx.x] += red[threadIdx.x + s];
    __syncthreads();
  }
  if (threadIdx.x == 0) atomicAdd(acc, red[0]);
}

// con2: per-user dot products (wave per user), permU ~ u*48271+7 mod U.
__global__ __launch_bounds__(32) void con2_kernel(
    const float* __restrict__ ue, const float* __restrict__ uec,
    float* __restrict__ acc) {
  int u = blockIdx.x;
  if (u >= U_) return;
  int lane = threadIdx.x;
  int up = (int)(((long long)u * 48271LL + 7LL) % U_);
  float s1 = 0.0f, s2 = 0.0f;
#pragma unroll
  for (int j = lane; j < H_; j += 32) {
    float a = ue[(size_t)u * H_ + j];
    s1 += a * uec[(size_t)u * H_ + j];
    s2 += a * uec[(size_t)up * H_ + j];
  }
  s1 = wave_sum(s1);
  s2 = wave_sum(s2);
  if (lane == 0) {
    float pos = s1 * (1.0f / H_), neg = s2 * (1.0f / H_);
    atomicAdd(acc, -logf(1e-8f + sigmoidf(pos)) -
                       logf(1e-8f + 1.0f - sigmoidf(neg)));
  }
}

// log_feats = E_b + LN(Fu[b,l]); logits = dot with raw item_emb rows.
__global__ __launch_bounds__(256) void logits_kernel(
    const float* __restrict__ Fu, const float* __restrict__ E,
    const float* __restrict__ ln_g, const float* __restrict__ ln_b,
    const float* __restrict__ item_emb_raw, const int* __restrict__ pos_seqs,
    const int* __restrict__ neg_seqs, float* __restrict__ pos_out,
    float* __restrict__ neg_out) {
  int w = blockIdx.x * 8 + (threadIdx.x >> 5);
  if (w >= B_ * L_) return;
  const int lane = threadIdx.x & 31;
  const int b = w / L_;
  const float* fr = Fu + (size_t)w * H_;
  float v[4], s = 0.0f;
#pragma unroll
  for (int j = 0; j < 4; ++j) { v[j] = fr[lane + 32 * j]; s += v[j]; }
  float mean = wave_sum(s) * (1.0f / H_);
  float vs = 0.0f;
#pragma unroll
  for (int j = 0; j < 4; ++j) { float d = v[j] - mean; vs += d * d; }
  float inv = rsqrtf(wave_sum(vs) * (1.0f / H_) + 1e-8f);
  const float* Eb = E + (size_t)b * H_;
  int pid = pos_seqs[w], nid = neg_seqs[w];
  float dp = 0.0f, dn = 0.0f;
#pragma unroll
  for (int j = 0; j < 4; ++j) {
    int h = lane + 32 * j;
    float lf = (v[j] - mean) * inv * ln_g[h] + ln_b[h] + Eb[h];
    dp += lf * item_emb_raw[(size_t)pid * H_ + h];
    dn += lf * item_emb_raw[(size_t)nid * H_ + h];
  }
  dp = wave_sum(dp);
  dn = wave_sum(dn);
  if (lane == 0) { pos_out[w] = dp; neg_out[w] = dn; }
}

__global__ void loss_kernel(const float* __restrict__ con1_acc,
                            const float* __restrict__ con2_acc,
                            float* __restrict__ out) {
  if (threadIdx.x == 0 && blockIdx.x == 0) {
    float con1 = con1_acc[0] * (1.0f / (B_ * H_));
    float con2 = con2_acc[0] * (1.0f / U_);
    out[0] = 0.5f * con1 + 0.5f * con2;   // BETA = 0.5
  }
}

// ---------------------------------------------------------------------------
// Orchestration
// ---------------------------------------------------------------------------
extern "C" void kernel_launch(void* const* d_in, const int* in_sizes, int n_in,
                              void* d_out, int out_size, void* d_ws,
                              size_t ws_size, hipStream_t stream) {
  (void)in_sizes; (void)n_in; (void)out_size; (void)ws_size;

  const float* p_user_emb = (const float*)d_in[0];
  const float* p_item_emb = (const float*)d_in[1];
  const float* p_cate_emb = (const float*)d_in[2];
  const float* p_year_emb = (const float*)d_in[3];
  const float* p_month_emb= (const float*)d_in[4];
  const float* p_day_emb  = (const float*)d_in[5];
  const float* p_abs_pos  = (const float*)d_in[6];
  // d_in[7] mu_all, d_in[8] sigma_all: dead (fi collapses to uniform)
  const float* p_gcW  = (const float*)d_in[9];
  const float* p_gcb  = (const float*)d_in[10];
  const float* p_biW  = (const float*)d_in[11];
  const float* p_bib  = (const float*)d_in[12];
  const float* p_gcWc = (const float*)d_in[13];
  const float* p_gcbc = (const float*)d_in[14];
  const float* p_biWc = (const float*)d_in[15];
  const float* p_bibc = (const float*)d_in[16];
  const float* p_Wih  = (const float*)d_in[17];
  const float* p_Whh  = (const float*)d_in[18];
  const float* p_bih  = (const float*)d_in[19];
  const float* p_bhh  = (const float*)d_in[20];
  const float* p_lng  = (const float*)d_in[21];
  const float* p_lnb  = (const float*)d_in[22];
  const int*   ui_rows = (const int*)d_in[23];
  const int*   ui_cols = (const int*)d_in[24];
  const float* ui_vals = (const float*)d_in[25];
  const int*   uc_rows = (const int*)d_in[26];
  const int*   uc_cols = (const int*)d_in[27];
  const float* uc_vals = (const float*)d_in[28];
  // d_in[29] it_rows implicit (repeat(arange(I),3))
  const int*   it_cols = (const int*)d_in[30];
  const float* it_vals = (const float*)d_in[31];
  // d_in[32] time_int: dead
  const int*   user_ids = (const int*)d_in[33];
  const int*   log_seqs = (const int*)d_in[34];
  // d_in[35..37] year/month/day: unused in forward
  const int*   pos_seqs = (const int*)d_in[38];
  const int*   neg_seqs = (const int*)d_in[39];

  float* out = (float*)d_out;
  float* pos_out  = out;                 // [B,L]
  float* neg_out  = out + B_ * L_;       // [B,L]
  float* loss_out = out + 2 * B_ * L_;   // scalar

  // --- workspace layout (floats) ---
  float* ws = (float*)d_ws;
  long long off = 0;
  float* all_ui    = ws + off; off += (long long)NUI * H_;
  float* all_uc    = ws + off; off += (long long)NUC * H_;
  float* item_time = ws + off; off += (long long)I_ * H_;
  float* Fu        = ws + off; off += (long long)B_ * L_ * H_;
  float* seq_mean  = ws + off; off += B_ * H_;
  float* Emat      = ws + off; off += B_ * H_;
  float* flm       = ws + off; off += H_;
  float* accs      = ws + off; off += 64;        // [0]=con1, [1]=con2
  // f16 weight mirrors (converted once per launch)
  _Float16* wh     = (_Float16*)(ws + off);
  long long hoff = 0;
  _Float16* gcW16  = wh + hoff; hoff += (long long)NL_ * H_ * H_;
  _Float16* biW16  = wh + hoff; hoff += (long long)NL_ * H_ * H_;
  _Float16* gcWc16 = wh + hoff; hoff += (long long)NL_ * H_ * H_;
  _Float16* biWc16 = wh + hoff; hoff += (long long)NL_ * H_ * H_;
  _Float16* Wih16  = wh + hoff; hoff += 3LL * H_ * H_;
  _Float16* Whh16  = wh + hoff; hoff += 3LL * H_ * H_;
  off += (hoff + 1) / 2;
  float* scratch   = ws + off;                   // time-shared region
  float* ego  = scratch;                         // NUI*H
  float* side = scratch + (long long)NUI * H_;   // NUI*H
  float* accN = scratch + 2LL * (long long)NUI * H_;  // NUI*H
  float* seqs = scratch;                              // B*L*H (after NGCF)
  float* xp   = scratch + (long long)B_ * L_ * H_;    // B*L*384

  auto g1 = [](long long n) { return (unsigned)((n + 255) / 256); };

  zero_kernel<<<1, 256, 0, stream>>>(accs, 64);

  // f16 weight conversion (tiny; done every launch for determinism)
  const long long WN = (long long)NL_ * H_ * H_;
  cvt_f16_kernel<<<g1(WN), 256, 0, stream>>>(p_gcW,  gcW16,  WN);
  cvt_f16_kernel<<<g1(WN), 256, 0, stream>>>(p_biW,  biW16,  WN);
  cvt_f16_kernel<<<g1(WN), 256, 0, stream>>>(p_gcWc, gcWc16, WN);
  cvt_f16_kernel<<<g1(WN), 256, 0, stream>>>(p_biWc, biWc16, WN);
  cvt_f16_kernel<<<g1(WN), 256, 0, stream>>>(p_Wih,  Wih16,  WN);
  cvt_f16_kernel<<<g1(WN), 256, 0, stream>>>(p_Whh,  Whh16,  WN);

  // ================= NGCF on user-item graph =================
  {
    const long long NE = (long long)NUI * H_;
    concat2_kernel<<<g1(NE), 256, 0, stream>>>(p_user_emb, U_, p_item_emb, I_,
                                               ego, accN);
    for (int i = 0; i < NL_; ++i) {
      zero_kernel<<<g1(NE), 256, 0, stream>>>(side, NE);
      spmm_scatter_kernel<<<g1((long long)NNZ_UI_ * 32), 256, 0, stream>>>(
          ui_rows, ui_cols, ui_vals, ego, side, NNZ_UI_);
      ngcf_dense_kernel<<<(NUI + 15) / 16, 32, 0, stream>>>(
          ego, side, ego, accN, gcW16 + (size_t)i * H_ * H_, p_gcb + i * H_,
          biW16 + (size_t)i * H_ * H_, p_bib + i * H_, NUI);
    }
    scale_kernel<<<g1(NE), 256, 0, stream>>>(accN, all_ui, NE,
                                             1.0f / (NL_ + 1));
  }

  // ================= NGCF on user-cate graph =================
  {
    const long long NE = (long long)NUC * H_;
    concat2_kernel<<<g1(NE), 256, 0, stream>>>(p_user_emb, U_, p_cate_emb, C_,
                                               ego, accN);
    for (int i = 0; i < NL_; ++i) {
      zero_kernel<<<g1(NE), 256, 0, stream>>>(side, NE);
      spmm_scatter_kernel<<<g1((long long)NNZ_UC_ * 32), 256, 0, stream>>>(
          uc_rows, uc_cols, uc_vals, ego, side, NNZ_UC_);
      ngcf_dense_kernel<<<(NUC + 15) / 16, 32, 0, stream>>>(
          ego, side, ego, accN, gcWc16 + (size_t)i * H_ * H_, p_gcbc + i * H_,
          biWc16 + (size_t)i * H_ * H_, p_bibc + i * H_, NUC);
    }
    scale_kernel<<<g1(NE), 256, 0, stream>>>(accN, all_uc, NE,
                                             1.0f / (NL_ + 1));
  }

  const float* ngcf_user  = all_ui;                    // all_ui[:U]
  const float* ngcf_items = all_ui + (size_t)U_ * H_;  // all_ui[U:]

  con2_kernel<<<U_, 32, 0, stream>>>(ngcf_user, all_uc, accs + 1);

  // Sequence features
  item_time_kernel<<<g1((long long)I_ * H_), 256, 0, stream>>>(
      it_cols, it_vals, p_year_emb, p_month_emb, p_day_emb, item_time);
  seqs_kernel<<<g1((long long)B_ * L_ * H_), 256, 0, stream>>>(
      log_seqs, ngcf_items, p_abs_pos, item_time, seqs);
  seq_mean_kernel<<<g1(B_ * H_), 256, 0, stream>>>(seqs, seq_mean);
  ln_rows_kernel<<<B_, 32, 0, stream>>>(seq_mean, p_lng, p_lnb, Emat, B_);

  // GRU
  xp_gemm_kernel<<<(B_ * L_) / 16, 32, 0, stream>>>(seqs, Wih16, p_bih, xp);
  const unsigned gru_lds = WHH_HALVES * 2 + 16 * 132 * 4 + 16 * 384 * 4;
  gru_scan_kernel<<<B_ / 16, 32, gru_lds, stream>>>(xp, Whh16, p_bhh, Fu);
  flmean_kernel<<<1, 128, 0, stream>>>(Fu, flm);

  // Losses + logits
  con1_kernel<<<g1(B_ * H_), 256, 0, stream>>>(ngcf_user, user_ids, flm, accs);
  logits_kernel<<<(B_ * L_ + 7) / 8, 256, 0, stream>>>(
      Fu, Emat, p_lng, p_lnb, p_item_emb, pos_seqs, neg_seqs, pos_out, neg_out);
  loss_kernel<<<1, 32, 0, stream>>>(accs, accs + 1, loss_out);
}
